// BondUpdateLayer_16020228014617
// MI455X (gfx1250) — matrix operand order repaired and checked by reading, not compiled, not run on hardware
//
#include <hip/hip_runtime.h>
#include <math.h>

// ---------------------------------------------------------------------------
// BondUpdateLayer for MI455X (gfx1250, wave32, WMMA 16x16x32 f16->f32)
//
// Pipeline:
//  1) per-edge Gaussian r_feat [E,20] (f32) + h_bond -> f16
//  2) per-triplet kv feature build [T,192] f16 (181 real + 11 zero pad)
//  3) MLPs via v_wmma_f32_16x16x32_f16:
//       GEMM1 fused bias+LayerNorm+ReLU -> hdn f16
//       GEMM2 fused bias -> k/v/q f16
//  4) per-edge one-pass online-softmax attention (segments of idx_ji are
//     contiguous since idx_ji is sorted) -> out f32 [E,128]
// ---------------------------------------------------------------------------

typedef __attribute__((ext_vector_type(16))) _Float16 v16h;
typedef __attribute__((ext_vector_type(8)))  float    v8f;
typedef _Float16 half_t;

#define DHID   128
#define NGAUSS 20
#define NANG   13
#define KV_PAD 192   // 181 padded up to multiple of 32

// --- pack a [Ksrc,128] f32 weight into WMMA B-fragment order -----------------
// B layout (16-bit, 32x16 per tile): lane L, n = L%16, base = (L<16)?0:8,
// halfs[i] = B[kt*32 + base + (i&7) + (i>>3)*16][nt*16 + n].  Packed so each lane reads 32
// contiguous bytes:  dst[((nt*Ktiles + kt)*32 + L)*16 + i]
__global__ void pack_w_kernel(const float* __restrict__ W, half_t* __restrict__ dst,
                              int Ksrc, int Ktiles) {
    int idx = blockIdx.x * 256 + threadIdx.x;
    int total = Ktiles * 8 * 32 * 16;
    if (idx >= total) return;
    int i  = idx & 15;
    int L  = (idx >> 4) & 31;
    int kt = (idx >> 9) % Ktiles;
    int nt = (idx >> 9) / Ktiles;
    int n    = L & 15;
    int base = (L < 16) ? 0 : 8;
    int r = kt * 32 + base + (i & 7) + ((i >> 3) << 4);
    int c = nt * 16 + n;
    dst[idx] = (r < Ksrc) ? (half_t)W[(size_t)r * DHID + c] : (half_t)0.f;
}

// --- per-edge Gaussian smearing ---------------------------------------------
__global__ void edge_rfeat_kernel(const float* __restrict__ pos,
                                  const int* __restrict__ brow,
                                  const int* __restrict__ bcol,
                                  float* __restrict__ rfeat, int E) {
    int e = blockIdx.x * 256 + threadIdx.x;
    if (e >= E) return;
    int r = brow[e], c = bcol[e];
    float dx = pos[c*3+0] - pos[r*3+0];
    float dy = pos[c*3+1] - pos[r*3+1];
    float dz = pos[c*3+2] - pos[r*3+2];
    float d = sqrtf(dx*dx + dy*dy + dz*dz);
    const float step  = 10.0f / 19.0f;           // G_STOP/(NG-1)
    const float coeff = -0.5f / (step * step);
    #pragma unroll
    for (int g = 0; g < NGAUSS; ++g) {
        float t = d - step * (float)g;
        rfeat[(size_t)e * NGAUSS + g] = __expf(coeff * t * t);
    }
}

// --- f32 -> f16 bulk convert -------------------------------------------------
__global__ void cvt16_kernel(const float* __restrict__ src, half_t* __restrict__ dst,
                             size_t n) {
    size_t i = (size_t)blockIdx.x * 256 + threadIdx.x;
    if (i < n) dst[i] = (half_t)src[i];
}

// --- build kv feature rows [T,192] f16 --------------------------------------
// layout per row: [ h_bond[idx_kj] (128) | r_feat[idx_kj] (20) |
//                   r_feat[idx_ji] (20)  | angular (13) | zeros (11) ]
__global__ void triplet_kv_kernel(const float* __restrict__ pos,
                                  const half_t* __restrict__ hb16,
                                  const float* __restrict__ rfeat,
                                  const int* __restrict__ ii, const int* __restrict__ ij,
                                  const int* __restrict__ ik, const int* __restrict__ ikj,
                                  const int* __restrict__ iji,
                                  half_t* __restrict__ kv, int T) {
    int t   = blockIdx.x;
    int tid = threadIdx.x;      // 64 threads
    if (t >= T) return;
    int kj = ikj[t], ji = iji[t];
    half_t* row = kv + (size_t)t * KV_PAD;
    // 128 halfs of h_bond[kj]: 2 halfs per thread as one dword
    ((unsigned int*)row)[tid] = ((const unsigned int*)(hb16 + (size_t)kj * DHID))[tid];
    if (tid < NGAUSS) {
        row[DHID + tid] = (half_t)rfeat[(size_t)kj * NGAUSS + tid];
    } else if (tid < 2 * NGAUSS) {
        row[DHID + tid] = (half_t)rfeat[(size_t)ji * NGAUSS + (tid - NGAUSS)];
    } else if (tid < 2 * NGAUSS + NANG) {
        int f = tid - 2 * NGAUSS;
        int ai = ii[t], aj = ij[t], ak = ik[t];
        float pix = pos[ai*3], piy = pos[ai*3+1], piz = pos[ai*3+2];
        float jx = pos[aj*3]-pix, jy = pos[aj*3+1]-piy, jz = pos[aj*3+2]-piz;
        float kx = pos[ak*3]-pix, ky = pos[ak*3+1]-piy, kz = pos[ak*3+2]-piz;
        float dt = jx*kx + jy*ky + jz*kz;
        float cx = jy*kz - jz*ky, cy = jz*kx - jx*kz, cz = jx*ky - jy*kx;
        float cr = sqrtf(cx*cx + cy*cy + cz*cz);
        float ang = atan2f(cr, dt);
        const float freqs[6] = {1.f, 2.f, 3.f, 1.f, 0.5f, 1.f/3.f};
        float val;
        if (f == 0)      val = ang;
        else if (f < 7)  val = __sinf(ang * freqs[f-1]);
        else             val = __cosf(ang * freqs[f-7]);
        row[DHID + 2*NGAUSS + f] = (half_t)val;
    } else {
        // pad columns 181..191 with zeros
        row[DHID + 2*NGAUSS + NANG + (tid - (2*NGAUSS + NANG))] = (half_t)0.f;
    }
}

// --- WMMA GEMM: C[M,128] = A[M,Kpad] * Bpacked, fused epilogue --------------
// One wave owns a 16x128 row-stripe (8 accumulator tiles), A fragment reused
// across all 8 N-tiles.  do_ln_relu: 1 = bias+LayerNorm+ReLU, 0 = bias only.
// Output is f16 row-major [M,128] (M padded to 128; padded rows are junk but
// never consumed).
__global__ void wmma_mlp_kernel(const half_t* __restrict__ A, int Apitch, int Ktiles,
                                const half_t* __restrict__ Bp,
                                const float* __restrict__ bias,
                                const float* __restrict__ g, const float* __restrict__ bln,
                                half_t* __restrict__ out, int do_ln_relu) {
    int wave = threadIdx.x >> 5;
    int lane = threadIdx.x & 31;
    int mtile = blockIdx.x * 8 + wave;
    int m       = lane & 15;
    int hi_half = lane >> 4;                      // 0 or 1
    // A-fragment layout (16-bit 16x32): lanes<16: halfs = A[m][k..k+7],A[m][k+16..k+23]
    //                                   lanes>=16: +8 shift
    const half_t* Abase = A + (size_t)(mtile * 16 + m) * Apitch + hi_half * 8;

    v8f acc[8] = {};
    for (int kt = 0; kt < Ktiles; ++kt) {
        union { v16h v; uint4 q[2]; } ua;
        ua.q[0] = *(const uint4*)(Abase + kt * 32);
        ua.q[1] = *(const uint4*)(Abase + kt * 32 + 16);
        #pragma unroll
        for (int nt = 0; nt < 8; ++nt) {
            union { v16h v; uint4 q[2]; } ub;
            const uint4* bp =
                (const uint4*)(Bp + (size_t)(((nt * Ktiles + kt) * 32) + lane) * 16);
            ub.q[0] = bp[0];
            ub.q[1] = bp[1];
            acc[nt] = __builtin_amdgcn_wmma_f32_16x16x32_f16(
                false, ua.v, false, ub.v, (short)0, acc[nt], false, false);
        }
    }

    int n = lane & 15;
    #pragma unroll
    for (int nt = 0; nt < 8; ++nt) {
        float bb = bias[nt * 16 + n];
        #pragma unroll
        for (int r = 0; r < 8; ++r) acc[nt][r] += bb;
    }

    if (do_ln_relu) {
        #pragma unroll
        for (int r = 0; r < 8; ++r) {
            float s1 = 0.f, s2 = 0.f;
            #pragma unroll
            for (int nt = 0; nt < 8; ++nt) { float x = acc[nt][r]; s1 += x; s2 += x * x; }
            // row lives across one 16-lane half-wave
            #pragma unroll
            for (int msk = 1; msk < 16; msk <<= 1) {
                s1 += __shfl_xor(s1, msk, 16);
                s2 += __shfl_xor(s2, msk, 16);
            }
            float mu   = s1 * (1.f / 128.f);
            float var  = s2 * (1.f / 128.f) - mu * mu;
            float rstd = rsqrtf(var + 1e-5f);
            size_t grow = (size_t)mtile * 16 + r + hi_half * 8;
            #pragma unroll
            for (int nt = 0; nt < 8; ++nt) {
                int col = nt * 16 + n;
                float x = (acc[nt][r] - mu) * rstd * g[col] + bln[col];
                out[grow * DHID + col] = (half_t)fmaxf(x, 0.f);
            }
        }
    } else {
        #pragma unroll
        for (int r = 0; r < 8; ++r) {
            size_t grow = (size_t)mtile * 16 + r + hi_half * 8;
            #pragma unroll
            for (int nt = 0; nt < 8; ++nt)
                out[grow * DHID + nt * 16 + n] = (half_t)acc[nt][r];
        }
    }
}

// --- per-edge online-softmax attention --------------------------------------
// idx_ji is sorted -> each edge's triplets are a contiguous [start,end) range.
// 128 threads per edge: tid = h*8 + d (H=16, HD=8).  8-lane shuffles reduce
// the per-head q.k dot; flash-style single pass for softmax + weighted sum.
__global__ void attn_kernel(const half_t* __restrict__ qm, const half_t* __restrict__ km,
                            const half_t* __restrict__ vm, const int* __restrict__ iji,
                            float* __restrict__ out, int E, int T) {
    int e   = blockIdx.x;
    int tid = threadIdx.x;     // 128
    // lower_bound(idx_ji, e)
    int lo = 0, hi = T;
    while (lo < hi) { int mid = (lo + hi) >> 1; if (iji[mid] <  e) lo = mid + 1; else hi = mid; }
    int start = lo;
    hi = T;
    while (lo < hi) { int mid = (lo + hi) >> 1; if (iji[mid] <= e) lo = mid + 1; else hi = mid; }
    int end = lo;

    float qv = (float)qm[(size_t)e * DHID + tid];
    const float scale = 0.35355339059327373f;   // 1/sqrt(HD)
    float mx = -INFINITY, s = 0.f, acc = 0.f;
    for (int t = start; t < end; ++t) {
        float p = qv * (float)km[(size_t)t * DHID + tid];
        p += __shfl_xor(p, 1, 8);
        p += __shfl_xor(p, 2, 8);
        p += __shfl_xor(p, 4, 8);
        float l  = p * scale;
        float nm = fmaxf(mx, l);
        float c0 = __expf(mx - nm);
        float c1 = __expf(l - nm);
        s   = s   * c0 + c1;
        acc = acc * c0 + c1 * (float)vm[(size_t)t * DHID + tid];
        mx = nm;
    }
    out[(size_t)e * DHID + tid] = (end > start) ? acc / s : 0.f;
}

// ---------------------------------------------------------------------------
extern "C" void kernel_launch(void* const* d_in, const int* in_sizes, int n_in,
                              void* d_out, int out_size, void* d_ws, size_t ws_size,
                              hipStream_t stream) {
    (void)n_in; (void)out_size; (void)ws_size;
    const float* h_bond = (const float*)d_in[1];
    const float* pos    = (const float*)d_in[2];
    const float* hk_W1 = (const float*)d_in[3];
    const float* hk_b1 = (const float*)d_in[4];
    const float* hk_g  = (const float*)d_in[5];
    const float* hk_b  = (const float*)d_in[6];
    const float* hk_W2 = (const float*)d_in[7];
    const float* hk_b2 = (const float*)d_in[8];
    const float* hv_W1 = (const float*)d_in[9];
    const float* hv_b1 = (const float*)d_in[10];
    const float* hv_g  = (const float*)d_in[11];
    const float* hv_b  = (const float*)d_in[12];
    const float* hv_W2 = (const float*)d_in[13];
    const float* hv_b2 = (const float*)d_in[14];
    const float* hq_W1 = (const float*)d_in[15];
    const float* hq_b1 = (const float*)d_in[16];
    const float* hq_g  = (const float*)d_in[17];
    const float* hq_b  = (const float*)d_in[18];
    const float* hq_W2 = (const float*)d_in[19];
    const float* hq_b2 = (const float*)d_in[20];
    const int* brow = (const int*)d_in[21];
    const int* bcol = (const int*)d_in[22];
    const int* ii   = (const int*)d_in[23];
    const int* ij   = (const int*)d_in[24];
    const int* ik   = (const int*)d_in[25];
    const int* ikj  = (const int*)d_in[26];
    const int* iji  = (const int*)d_in[27];

    int E = in_sizes[21];
    int T = in_sizes[23];
    size_t Tpad = ((size_t)T + 127) & ~(size_t)127;
    size_t Epad = ((size_t)E + 127) & ~(size_t)127;
    size_t Hpad = Tpad > Epad ? Tpad : Epad;

    char* p = (char*)d_ws;
    auto alloc = [&](size_t bytes) {
        char* r = p;
        p += (bytes + 255) & ~(size_t)255;
        return r;
    };
    float*  rfeat = (float*) alloc(Epad * NGAUSS * sizeof(float));
    half_t* hb16  = (half_t*)alloc(Epad * DHID * sizeof(half_t));
    half_t* kv    = (half_t*)alloc(Tpad * KV_PAD * sizeof(half_t));
    half_t* hdn   = (half_t*)alloc(Hpad * DHID * sizeof(half_t));
    half_t* kf    = (half_t*)alloc(Tpad * DHID * sizeof(half_t));
    half_t* vf    = (half_t*)alloc(Tpad * DHID * sizeof(half_t));
    half_t* qf    = (half_t*)alloc(Epad * DHID * sizeof(half_t));
    const int KT1 = 6, KT2 = 4;          // 192/32 and 128/32 K-tiles
    const int PACK1 = KT1 * 8 * 32 * 16; // 24576 halfs
    const int PACK2 = KT2 * 8 * 32 * 16; // 16384 halfs
    half_t* wk1 = (half_t*)alloc(PACK1 * sizeof(half_t));
    half_t* wv1 = (half_t*)alloc(PACK1 * sizeof(half_t));
    half_t* wq1 = (half_t*)alloc(PACK2 * sizeof(half_t));
    half_t* wk2 = (half_t*)alloc(PACK2 * sizeof(half_t));
    half_t* wv2 = (half_t*)alloc(PACK2 * sizeof(half_t));
    half_t* wq2 = (half_t*)alloc(PACK2 * sizeof(half_t));

    // weight packing
    pack_w_kernel<<<(PACK1 + 255) / 256, 256, 0, stream>>>(hk_W1, wk1, 181, KT1);
    pack_w_kernel<<<(PACK1 + 255) / 256, 256, 0, stream>>>(hv_W1, wv1, 181, KT1);
    pack_w_kernel<<<(PACK2 + 255) / 256, 256, 0, stream>>>(hq_W1, wq1, 128, KT2);
    pack_w_kernel<<<(PACK2 + 255) / 256, 256, 0, stream>>>(hk_W2, wk2, 128, KT2);
    pack_w_kernel<<<(PACK2 + 255) / 256, 256, 0, stream>>>(hv_W2, wv2, 128, KT2);
    pack_w_kernel<<<(PACK2 + 255) / 256, 256, 0, stream>>>(hq_W2, wq2, 128, KT2);

    // features
    edge_rfeat_kernel<<<(E + 255) / 256, 256, 0, stream>>>(pos, brow, bcol, rfeat, E);
    cvt16_kernel<<<(unsigned)(((size_t)E * DHID + 255) / 256), 256, 0, stream>>>(
        h_bond, hb16, (size_t)E * DHID);
    triplet_kv_kernel<<<T, 64, 0, stream>>>(pos, hb16, rfeat, ii, ij, ik, ikj, iji, kv, T);

    // q path (per edge, not per triplet: 7x fewer MLP rows)
    wmma_mlp_kernel<<<(unsigned)(Epad / 128), 256, 0, stream>>>(
        hb16, DHID, KT2, wq1, hq_b1, hq_g, hq_b, hdn, 1);
    wmma_mlp_kernel<<<(unsigned)(Epad / 128), 256, 0, stream>>>(
        hdn, DHID, KT2, wq2, hq_b2, nullptr, nullptr, qf, 0);
    // k path
    wmma_mlp_kernel<<<(unsigned)(Tpad / 128), 256, 0, stream>>>(
        kv, KV_PAD, KT1, wk1, hk_b1, hk_g, hk_b, hdn, 1);
    wmma_mlp_kernel<<<(unsigned)(Tpad / 128), 256, 0, stream>>>(
        hdn, DHID, KT2, wk2, hk_b2, nullptr, nullptr, kf, 0);
    // v path
    wmma_mlp_kernel<<<(unsigned)(Tpad / 128), 256, 0, stream>>>(
        kv, KV_PAD, KT1, wv1, hv_b1, hv_g, hv_b, hdn, 1);
    wmma_mlp_kernel<<<(unsigned)(Tpad / 128), 256, 0, stream>>>(
        hdn, DHID, KT2, wv2, hv_b2, nullptr, nullptr, vf, 0);

    // attention + segment softmax
    attn_kernel<<<E, 128, 0, stream>>>(qf, kf, vf, iji, (float*)d_out, E, T);
}
